// ssa_55508157334017
// MI455X (gfx1250) — matrix operand order repaired and checked by reading, not compiled, not run on hardware
//
#include <hip/hip_runtime.h>

typedef __attribute__((ext_vector_type(16))) _Float16 v16h;
typedef __attribute__((ext_vector_type(8)))  _Float16 v8h;
typedef __attribute__((ext_vector_type(8)))  float    v8f;

#define B_   4
#define H_   128
#define W_   128
#define NPIX (H_ * W_)   /* 16384 */
#define CMID 128
#define CIN  128
#define CX   64
#define LD   40          /* LDS row stride in halves: 80B, 16B-aligned */

__device__ __forceinline__ v16h pack16(v8h lo, v8h hi) {
  union { v16h v; v8h h[2]; } u;
  u.h[0] = lo; u.h[1] = hi;
  return u.v;
}

// A fragment: 16x32 f16 (MxK). LDS holds [m][k] row-major, stride LD halves.
// Lane L: m = L&15; lanes 0-15 cover K {0..7,16..23}, lanes 16-31 {8..15,24..31}.
__device__ __forceinline__ v16h frag_a(const _Float16* base) {
  const int lane = threadIdx.x & 31;
  const _Float16* row = base + (lane & 15) * LD + ((lane >> 4) * 8);
  v8h lo = *(const v8h*)(row);
  v8h hi = *(const v8h*)(row + 16);
  return pack16(lo, hi);
}

// B fragment: 32x16 f16 (KxN), stored transposed in LDS as [n][k], stride LD.
// Lane L: n = L&15; lanes 0-15 cover K {0..7,16..23}, lanes 16-31 {8..15,24..31}.
__device__ __forceinline__ v16h frag_bt(const _Float16* base) {
  const int lane = threadIdx.x & 31;
  const _Float16* row = base + (lane & 15) * LD + ((lane >> 4) * 8);
  v8h lo = *(const v8h*)(row);
  v8h hi = *(const v8h*)(row + 16);
  return pack16(lo, hi);
}

__device__ __forceinline__ v8f wmma_f16(v16h a, v16h b, v8f c) {
  return __builtin_amdgcn_wmma_f32_16x16x32_f16(false, a, false, b, (short)0, c,
                                                false, false);
}

__device__ __forceinline__ float leaky(float x) { return x >= 0.f ? x : 0.2f * x; }

// ---------------------------------------------------------------- pack concat
__global__ __launch_bounds__(256) void pack_cat(const float* __restrict__ x1,
                                                const float* __restrict__ x2,
                                                _Float16* __restrict__ cat_h) {
  const int idx = blockIdx.x * 256 + threadIdx.x;
  if (idx >= B_ * CIN * NPIX) return;
  const int n = idx % NPIX;
  const int c = (idx / NPIX) % CIN;
  const int b = idx / (NPIX * CIN);
  const float v = (c < CX) ? x1[((size_t)(b * CX + c)) * NPIX + n]
                           : x2[((size_t)(b * CX + (c - CX))) * NPIX + n];
  cat_h[idx] = (_Float16)v;
}

// --------------------------------------------------- implicit-GEMM convolution
// One WG per (batch, image row): 128 pixels x 128 out channels.
// M = pixels, N = channels, K = CIN*RS.  MODE: 0 = leaky->f16 out,
// 1 = (1x1 skip) write f32 feat, 2 = leaky then feat += (feat holds skip).
template <int RS, int PAD, int MODE>
__global__ __launch_bounds__(256) void conv_gemm(const _Float16* __restrict__ in_h,
                                                 const float* __restrict__ wgt,
                                                 const float* __restrict__ bias,
                                                 _Float16* __restrict__ out_h,
                                                 float* __restrict__ feat) {
  __shared__ __attribute__((aligned(32))) _Float16 shA[128 * LD];  // [pixel][k]
  __shared__ __attribute__((aligned(32))) _Float16 shB[128 * LD];  // [co][k]
  const int b = blockIdx.x / H_;
  const int y = blockIdx.x % H_;
  const int t = threadIdx.x;
  const int wv = t >> 5;
  const int lane = t & 31;
  const int K = CIN * RS;
  const int KB = K / 32;

  v8f acc[8];
#pragma unroll
  for (int j = 0; j < 8; ++j)
#pragma unroll
    for (int e = 0; e < 8; ++e) acc[j][e] = 0.f;

  for (int kb = 0; kb < KB; ++kb) {
    const int rs  = (RS == 9) ? (kb >> 2) : 0;
    const int ci0 = (kb & 3) * 32;
    const int r = rs / 3, s = rs % 3;
    // A tile: im2col 128 pixels x 32 k
    {
      const int kl = t >> 3;
      const int p0 = (t & 7) * 16;
      const int ci = ci0 + kl;
      const int yy = y + r - PAD;
      const bool yok = (yy >= 0) && (yy < H_);
      const _Float16* src = in_h + ((size_t)(b * CIN + ci)) * NPIX + yy * W_;
#pragma unroll
      for (int i = 0; i < 16; ++i) {
        const int xx = p0 + i + s - PAD;
        _Float16 v = (_Float16)0.f;
        if (yok && xx >= 0 && xx < W_) v = src[xx];
        shA[(p0 + i) * LD + kl] = v;
      }
    }
    // B tile: weights transposed [co][k]
    {
      const int co = t >> 1;
      const int kh = (t & 1) * 16;
#pragma unroll
      for (int i = 0; i < 16; ++i) {
        const int kl = kh + i;
        shB[co * LD + kl] = (_Float16)wgt[(size_t)co * K + (ci0 + kl) * RS + rs];
      }
    }
    __syncthreads();
    const v16h af = frag_a(shA + wv * 16 * LD);
#pragma unroll
    for (int j = 0; j < 8; ++j)
      acc[j] = wmma_f16(af, frag_bt(shB + j * 16 * LD), acc[j]);
    __syncthreads();
  }

  const int half = lane >> 4;
  const int px = y * W_ + wv * 16 + half * 8;  // 8 consecutive pixels per lane
#pragma unroll
  for (int j = 0; j < 8; ++j) {
    const int co = j * 16 + (lane & 15);
    const float bb = bias[co];
    const size_t base = ((size_t)(b * CMID + co)) * NPIX + px;
    if (MODE == 0) {
      v8h o;
#pragma unroll
      for (int v = 0; v < 8; ++v) o[v] = (_Float16)leaky(acc[j][v] + bb);
      *(v8h*)(out_h + base) = o;
    } else if (MODE == 1) {
      float4 f0 = make_float4(acc[j][0] + bb, acc[j][1] + bb, acc[j][2] + bb, acc[j][3] + bb);
      float4 f1 = make_float4(acc[j][4] + bb, acc[j][5] + bb, acc[j][6] + bb, acc[j][7] + bb);
      *(float4*)(feat + base) = f0;
      *(float4*)(feat + base + 4) = f1;
    } else {
      float4 f0 = *(const float4*)(feat + base);
      float4 f1 = *(const float4*)(feat + base + 4);
      f0.x += leaky(acc[j][0] + bb); f0.y += leaky(acc[j][1] + bb);
      f0.z += leaky(acc[j][2] + bb); f0.w += leaky(acc[j][3] + bb);
      f1.x += leaky(acc[j][4] + bb); f1.y += leaky(acc[j][5] + bb);
      f1.z += leaky(acc[j][6] + bb); f1.w += leaky(acc[j][7] + bb);
      *(float4*)(feat + base) = f0;
      *(float4*)(feat + base + 4) = f1;
    }
  }
}

// ------------------------------------------------------------- L1 row norm
__global__ __launch_bounds__(256) void l1norm(const float* __restrict__ feat,
                                              _Float16* __restrict__ vt) {
  __shared__ float red[256];
  const int row = blockIdx.x;  // b*128 + c
  const float* src = feat + (size_t)row * NPIX;
  float s = 0.f;
  for (int i = threadIdx.x; i < NPIX; i += 256) s += fabsf(src[i]);
  red[threadIdx.x] = s;
  __syncthreads();
  for (int st = 128; st > 0; st >>= 1) {
    if (threadIdx.x < st) red[threadIdx.x] += red[threadIdx.x + st];
    __syncthreads();
  }
  const float inv = 1.f / (1e-6f + red[0]);
  _Float16* dst = vt + (size_t)row * NPIX;
  for (int i = threadIdx.x; i < NPIX; i += 256) dst[i] = (_Float16)(src[i] * inv);
}

__global__ __launch_bounds__(256) void zero_f32(float* __restrict__ p, int n) {
  const int i = blockIdx.x * 256 + threadIdx.x;
  if (i < n) p[i] = 0.f;
}

// -------------------------------------------------- G = Vt * Vt^T  (K-split)
__global__ __launch_bounds__(256) void gram(const _Float16* __restrict__ vt,
                                            float* __restrict__ G) {
  __shared__ __attribute__((aligned(32))) _Float16 shV[128 * LD];  // [c][k]
  const int b = blockIdx.y;
  const int n0 = blockIdx.x * 256;  // 64 K-slices of 256
  const int t = threadIdx.x, wv = t >> 5, lane = t & 31;
  v8f acc[8];
#pragma unroll
  for (int j = 0; j < 8; ++j)
#pragma unroll
    for (int e = 0; e < 8; ++e) acc[j][e] = 0.f;

  for (int kb = 0; kb < 8; ++kb) {
    const int k0 = n0 + kb * 32;
    {
      const int c = t >> 1;
      const int kh = (t & 1) * 16;
      const _Float16* src = vt + ((size_t)(b * CMID + c)) * NPIX + k0 + kh;
      *(v8h*)(&shV[c * LD + kh])     = *(const v8h*)(src);
      *(v8h*)(&shV[c * LD + kh + 8]) = *(const v8h*)(src + 8);
    }
    __syncthreads();
    const v16h af = frag_a(shV + wv * 16 * LD);
#pragma unroll
    for (int j = 0; j < 8; ++j)
      acc[j] = wmma_f16(af, frag_bt(shV + j * 16 * LD), acc[j]);
    __syncthreads();
  }
  const int half = lane >> 4;
#pragma unroll
  for (int j = 0; j < 8; ++j) {
    const int d = j * 16 + (lane & 15);
#pragma unroll
    for (int v = 0; v < 8; ++v) {
      const int c = wv * 16 + v + 8 * half;
      atomicAdd(&G[(size_t)b * CMID * CMID + c * CMID + d], acc[j][v]);
    }
  }
}

// ----------------------------------------------- T = Vt * x1^T  (K-split)
__global__ __launch_bounds__(256) void tmat(const _Float16* __restrict__ vt,
                                            const _Float16* __restrict__ cat_h,
                                            float* __restrict__ T) {
  __shared__ __attribute__((aligned(32))) _Float16 shV[128 * LD];  // [c][k]
  __shared__ __attribute__((aligned(32))) _Float16 shX[64 * LD];   // [m][k]
  const int b = blockIdx.y;
  const int n0 = blockIdx.x * 256;
  const int t = threadIdx.x, wv = t >> 5, lane = t & 31;
  v8f acc[4];
#pragma unroll
  for (int j = 0; j < 4; ++j)
#pragma unroll
    for (int e = 0; e < 8; ++e) acc[j][e] = 0.f;

  for (int kb = 0; kb < 8; ++kb) {
    const int k0 = n0 + kb * 32;
    {
      const int c = t >> 1;
      const int kh = (t & 1) * 16;
      const _Float16* src = vt + ((size_t)(b * CMID + c)) * NPIX + k0 + kh;
      *(v8h*)(&shV[c * LD + kh])     = *(const v8h*)(src);
      *(v8h*)(&shV[c * LD + kh + 8]) = *(const v8h*)(src + 8);
    }
    if (t < 128) {  // x1 = first 64 channels of cat_h
      const int m = t >> 1;
      const int kh = (t & 1) * 16;
      const _Float16* src = cat_h + ((size_t)(b * CIN + m)) * NPIX + k0 + kh;
      *(v8h*)(&shX[m * LD + kh])     = *(const v8h*)(src);
      *(v8h*)(&shX[m * LD + kh + 8]) = *(const v8h*)(src + 8);
    }
    __syncthreads();
    const v16h af = frag_a(shV + wv * 16 * LD);
#pragma unroll
    for (int j = 0; j < 4; ++j)
      acc[j] = wmma_f16(af, frag_bt(shX + j * 16 * LD), acc[j]);
    __syncthreads();
  }
  const int half = lane >> 4;
#pragma unroll
  for (int j = 0; j < 4; ++j) {
    const int m = j * 16 + (lane & 15);
#pragma unroll
    for (int v = 0; v < 8; ++v) {
      const int c = wv * 16 + v + 8 * half;
      atomicAdd(&T[(size_t)b * CMID * CX + c * CX + m], acc[j][v]);
    }
  }
}

// ------------------------------------ in-place Gauss-Jordan 128x128 inverse
__global__ __launch_bounds__(256) void invert128(const float* __restrict__ G,
                                                 float* __restrict__ Ginv) {
  __shared__ float sM[128 * 128];  // 64 KB
  const int b = blockIdx.x;
  const int t = threadIdx.x;
  for (int i = t; i < 128 * 128; i += 256) sM[i] = G[(size_t)b * 128 * 128 + i];
  __syncthreads();
  for (int p = 0; p < 128; ++p) {
    const float piv = sM[p * 128 + p];
    const float d = 1.f / piv;
    __syncthreads();
    if (t < 128) sM[p * 128 + t] = (t == p) ? d : sM[p * 128 + t] * d;
    __syncthreads();
    if (t < 128 && t != p) {
      const float f = sM[t * 128 + p];
      for (int j = 0; j < 128; ++j) {
        const float pv = sM[p * 128 + j];
        sM[t * 128 + j] = (j == p) ? (-f * d) : (sM[t * 128 + j] - f * pv);
      }
    }
    __syncthreads();
  }
  for (int i = t; i < 128 * 128; i += 256) Ginv[(size_t)b * 128 * 128 + i] = sM[i];
}

// ------------------------------------------- T2 = Ginv * T, convert to f16
__global__ __launch_bounds__(256) void t2comp(const float* __restrict__ Ginv,
                                              const float* __restrict__ T,
                                              _Float16* __restrict__ t2h) {
  const int b = blockIdx.x;
  for (int idx = threadIdx.x; idx < CMID * CX; idx += 256) {
    const int c = idx >> 6, m = idx & 63;
    float a = 0.f;
    for (int d = 0; d < CMID; ++d)
      a += Ginv[(size_t)b * CMID * CMID + c * CMID + d] *
           T[(size_t)b * CMID * CX + d * CX + m];
    t2h[((size_t)(b * CMID + c)) * CX + m] = (_Float16)a;
  }
}

// --------------------------- bridge[b][m][n] = sum_c Vt[c][n] * T2[c][m]
__global__ __launch_bounds__(256) void projmm(const _Float16* __restrict__ vt,
                                              const _Float16* __restrict__ t2h,
                                              float* __restrict__ out) {
  __shared__ __attribute__((aligned(32))) _Float16 shP[128 * LD];  // [pixel][c]
  __shared__ __attribute__((aligned(32))) _Float16 shT[64 * LD];   // [m][c]
  const int b = blockIdx.y;
  const int n0 = blockIdx.x * 128;
  const int t = threadIdx.x, wv = t >> 5, lane = t & 31;
  v8f acc[4];
#pragma unroll
  for (int j = 0; j < 4; ++j)
#pragma unroll
    for (int e = 0; e < 8; ++e) acc[j][e] = 0.f;

  for (int kb = 0; kb < 4; ++kb) {
    const int c0 = kb * 32;
    {
      const int cl = t & 31;
      const int p0 = (t >> 5) * 16;
      const _Float16* src = vt + ((size_t)(b * CMID + c0 + cl)) * NPIX + n0 + p0;
#pragma unroll
      for (int i = 0; i < 16; ++i) shP[(p0 + i) * LD + cl] = src[i];
    }
    {
      const int m = t >> 2;
      const int ch = (t & 3) * 8;
#pragma unroll
      for (int i = 0; i < 8; ++i)
        shT[m * LD + ch + i] = t2h[((size_t)(b * CMID + c0 + ch + i)) * CX + m];
    }
    __syncthreads();
    const v16h af = frag_a(shP + wv * 16 * LD);
#pragma unroll
    for (int j = 0; j < 4; ++j)
      acc[j] = wmma_f16(af, frag_bt(shT + j * 16 * LD), acc[j]);
    __syncthreads();
  }
  const int half = lane >> 4;
#pragma unroll
  for (int j = 0; j < 4; ++j) {
    const int m = j * 16 + (lane & 15);
    const size_t base = ((size_t)(b * CX + m)) * NPIX + n0 + wv * 16 + half * 8;
    float4 f0 = make_float4(acc[j][0], acc[j][1], acc[j][2], acc[j][3]);
    float4 f1 = make_float4(acc[j][4], acc[j][5], acc[j][6], acc[j][7]);
    *(float4*)(out + base) = f0;
    *(float4*)(out + base + 4) = f1;
  }
}

// ---------------------------------------------------------------------------
extern "C" void kernel_launch(void* const* d_in, const int* in_sizes, int n_in,
                              void* d_out, int out_size, void* d_ws, size_t ws_size,
                              hipStream_t stream) {
  const float* x1 = (const float*)d_in[0];
  const float* x2 = (const float*)d_in[1];
  const float* w1 = (const float*)d_in[2];
  const float* b1 = (const float*)d_in[3];
  const float* w2 = (const float*)d_in[4];
  const float* b2 = (const float*)d_in[5];
  const float* wi = (const float*)d_in[6];
  const float* bi = (const float*)d_in[7];
  float* out = (float*)d_out;

  char* ws = (char*)d_ws;
  size_t off = 0;
  auto alloc = [&](size_t bytes) {
    size_t o = off;
    off = (off + bytes + 255) & ~(size_t)255;
    return o;
  };
  const size_t nCat = (size_t)B_ * CIN * NPIX;
  _Float16* cat_h = (_Float16*)(ws + alloc(nCat * 2));
  _Float16* h1    = (_Float16*)(ws + alloc(nCat * 2));
  float*    feat  = (float*)(ws + alloc(nCat * 4));
  _Float16* vt    = (_Float16*)(ws + alloc(nCat * 2));
  float*    G     = (float*)(ws + alloc((size_t)B_ * CMID * CMID * 4));
  float*    T     = (float*)(ws + alloc((size_t)B_ * CMID * CX * 4));
  float*    Ginv  = (float*)(ws + alloc((size_t)B_ * CMID * CMID * 4));
  _Float16* t2h   = (_Float16*)(ws + alloc((size_t)B_ * CMID * CX * 2));

  pack_cat<<<(int)((nCat + 255) / 256), 256, 0, stream>>>(x1, x2, cat_h);

  conv_gemm<9, 1, 0><<<B_ * H_, 256, 0, stream>>>(cat_h, w1, b1, h1, nullptr);
  conv_gemm<1, 0, 1><<<B_ * H_, 256, 0, stream>>>(cat_h, wi, bi, nullptr, feat);
  conv_gemm<9, 1, 2><<<B_ * H_, 256, 0, stream>>>(h1, w2, b2, nullptr, feat);

  l1norm<<<B_ * CMID, 256, 0, stream>>>(feat, vt);

  // G (65536 floats) and T (32768 floats) are contiguous: zero both.
  const int nz = B_ * CMID * CMID + B_ * CMID * CX;
  zero_f32<<<(nz + 255) / 256, 256, 0, stream>>>(G, nz);

  gram<<<dim3(64, B_), 256, 0, stream>>>(vt, G);
  tmat<<<dim3(64, B_), 256, 0, stream>>>(vt, cat_h, T);
  invert128<<<B_, 256, 0, stream>>>(G, Ginv);
  t2comp<<<B_, 256, 0, stream>>>(Ginv, T, t2h);
  projmm<<<dim3(128, B_), 256, 0, stream>>>(vt, t2h, out);
}
